// CPGTaintFlowModel_79242146611963
// MI455X (gfx1250) — matrix operand order repaired and checked by reading, not compiled, not run on hardware
//
#include <hip/hip_runtime.h>

#define HID   128
#define HEADS 4
#define HD    32

typedef __attribute__((ext_vector_type(16))) __bf16 v16bf;
typedef __attribute__((ext_vector_type(8)))  __bf16 v8bf;
typedef __attribute__((ext_vector_type(8)))  float  v8f;

// ---------- helpers ----------
__device__ __forceinline__ unsigned short bfbits(float f) {
  unsigned u = __float_as_uint(f);
  u += 0x7FFFu + ((u >> 16) & 1u);     // round-to-nearest-even
  return (unsigned short)(u >> 16);
}
__device__ __forceinline__ __bf16 f2bf(float f) {
  union { unsigned short s; __bf16 b; } cv; cv.s = bfbits(f); return cv.b;
}
// monotonic float<->uint mapping for atomicMax on floats
__device__ __forceinline__ unsigned encf(float f) {
  unsigned u = __float_as_uint(f);
  return (u & 0x80000000u) ? ~u : (u | 0x80000000u);
}
__device__ __forceinline__ float decf(unsigned u) {
  unsigned b = (u & 0x80000000u) ? (u & 0x7FFFFFFFu) : ~u;
  return __uint_as_float(b);
}

// ---------- packed f32 -> bf16 convert (4 elems/thread, n % 4 == 0) ----------
__global__ void cvt_bf16_kernel(const float* __restrict__ src, __bf16* __restrict__ dst,
                                long long n) {
  long long i = ((long long)blockIdx.x * blockDim.x + threadIdx.x) * 4;
  if (i >= n) return;
  float4 v = *(const float4*)(src + i);
  unsigned lo = ((unsigned)bfbits(v.y) << 16) | bfbits(v.x);
  unsigned hi = ((unsigned)bfbits(v.w) << 16) | bfbits(v.z);
  uint2 r; r.x = lo; r.y = hi;
  *(uint2*)(dst + i) = r;
}

// ---------- weight convert + transpose: W[K,Nc] f32 -> WT[Nc,K] bf16 ----------
__global__ void cvt_tr_kernel(const float* __restrict__ W, __bf16* __restrict__ WT,
                              int K, int Nc) {
  int idx = blockIdx.x * blockDim.x + threadIdx.x;
  if (idx >= K * Nc) return;
  int k = idx / Nc, n = idx - k * Nc;
  WT[(size_t)n * K + k] = f2bf(W[idx]);
}

// ---------- WMMA GEMM: C[M,Nc] = A[M,K] @ B[K,Nc] (+bias) ----------
// A: bf16 row-major [M,K]; BT: bf16 transposed weights [Nc,K]; C: f32.
// One wave computes a 16x64 tile: one shared A fragment, 4 accumulators.
// Requires M%16==0, Nc%64==0, K%32==0 (true for all shapes here).
__global__ void wmma_gemm_bf16(const __bf16* __restrict__ A, const __bf16* __restrict__ BT,
                               const float* __restrict__ bias, float* __restrict__ C,
                               int M, int Nc, int K) {
  const int wavesPerBlock = blockDim.x >> 5;
  int wave   = blockIdx.x * wavesPerBlock + (threadIdx.x >> 5);
  int tilesN = Nc >> 6;
  int tilesM = M >> 4;
  if (wave >= tilesM * tilesN) return;
  int tm = wave / tilesN, tn = wave % tilesN;
  int lane = threadIdx.x & 31;
  int half = lane >> 4, l16 = lane & 15;
  int row  = (tm << 4) + l16;
  int colBase = (tn << 6) + l16;

  const __bf16* Arow = A + (size_t)row * K;
  const __bf16* Bcol0 = BT + (size_t)(colBase)      * K;
  const __bf16* Bcol1 = BT + (size_t)(colBase + 16) * K;
  const __bf16* Bcol2 = BT + (size_t)(colBase + 32) * K;
  const __bf16* Bcol3 = BT + (size_t)(colBase + 48) * K;

  v8f acc0 = {}, acc1 = {}, acc2 = {}, acc3 = {};
  for (int k0 = 0; k0 < K; k0 += 32) {
    // A fragment (16x32): lanes 0-15 row l16, elems 0-15 = K k0..k0+15;
    // lanes 16-31 K k0+16..k0+31 (same k order as B)
    int ka = k0 + (half << 4);
    v8bf alo = *(const v8bf*)(Arow + ka);
    v8bf ahi = *(const v8bf*)(Arow + ka + 8);
    v16bf a = __builtin_shufflevector(alo, ahi, 0,1,2,3,4,5,6,7,8,9,10,11,12,13,14,15);
    // B fragment (32x16): lane col = l16; lanes 0-15 K=k0..k0+15, lanes 16-31 K=k0+16..k0+31.
    // Transposed weights make this one contiguous 16-element run per lane.
    int kb = k0 + (half << 4);
    v8bf b0lo = *(const v8bf*)(Bcol0 + kb), b0hi = *(const v8bf*)(Bcol0 + kb + 8);
    v8bf b1lo = *(const v8bf*)(Bcol1 + kb), b1hi = *(const v8bf*)(Bcol1 + kb + 8);
    v8bf b2lo = *(const v8bf*)(Bcol2 + kb), b2hi = *(const v8bf*)(Bcol2 + kb + 8);
    v8bf b3lo = *(const v8bf*)(Bcol3 + kb), b3hi = *(const v8bf*)(Bcol3 + kb + 8);
    v16bf b0 = __builtin_shufflevector(b0lo, b0hi, 0,1,2,3,4,5,6,7,8,9,10,11,12,13,14,15);
    v16bf b1 = __builtin_shufflevector(b1lo, b1hi, 0,1,2,3,4,5,6,7,8,9,10,11,12,13,14,15);
    v16bf b2 = __builtin_shufflevector(b2lo, b2hi, 0,1,2,3,4,5,6,7,8,9,10,11,12,13,14,15);
    v16bf b3 = __builtin_shufflevector(b3lo, b3hi, 0,1,2,3,4,5,6,7,8,9,10,11,12,13,14,15);
    acc0 = __builtin_amdgcn_wmma_f32_16x16x32_bf16(false, a, false, b0, (short)0, acc0, false, false);
    acc1 = __builtin_amdgcn_wmma_f32_16x16x32_bf16(false, a, false, b1, (short)0, acc1, false, false);
    acc2 = __builtin_amdgcn_wmma_f32_16x16x32_bf16(false, a, false, b2, (short)0, acc2, false, false);
    acc3 = __builtin_amdgcn_wmma_f32_16x16x32_bf16(false, a, false, b3, (short)0, acc3, false, false);
  }
  // C layout: lanes 0-15 VGPR j -> (M=tm*16+j, N=col); lanes 16-31 -> M=tm*16+8+j
  int rbase = (tm << 4) + (half << 3);
  float bb0 = bias ? bias[colBase]      : 0.f;
  float bb1 = bias ? bias[colBase + 16] : 0.f;
  float bb2 = bias ? bias[colBase + 32] : 0.f;
  float bb3 = bias ? bias[colBase + 48] : 0.f;
#pragma unroll
  for (int j = 0; j < 8; ++j) {
    float* Cr = C + (size_t)(rbase + j) * Nc + colBase;
    Cr[0]  = acc0[j] + bb0;
    Cr[16] = acc1[j] + bb1;
    Cr[32] = acc2[j] + bb2;
    Cr[48] = acc3[j] + bb3;
  }
}

// ---------- per-node attention logits: es/ed[n,h] = <h[n,h,:], a_s/a_d[h,:]> ----------
__global__ void node_dots(const float* __restrict__ h, const float* __restrict__ a_s,
                          const float* __restrict__ a_d, float* __restrict__ es,
                          float* __restrict__ ed, int N) {
  int idx = blockIdx.x * blockDim.x + threadIdx.x;  // n*HEADS + hh
  if (idx >= N * HEADS) return;
  int n = idx >> 2, hh = idx & 3;
  const float* hp = h + (size_t)n * (HEADS * HID) + hh * HID;
  const float* as = a_s + hh * HID;
  const float* ad = a_d + hh * HID;
  float s1 = 0.f, s2 = 0.f;
#pragma unroll 4
  for (int d = 0; d < HID; ++d) { float hv = hp[d]; s1 += hv * as[d]; s2 += hv * ad[d]; }
  es[idx] = s1; ed[idx] = s2;
}

// ---------- segment max over dst (edges + self loops), t<0 => all edges active ----------
__global__ void edge_max_kernel(const int* __restrict__ src, const int* __restrict__ dst,
                                const int* __restrict__ etyp, int t,
                                const float* __restrict__ es, const float* __restrict__ ed,
                                unsigned* __restrict__ menc, int E, int N) {
  int idx = blockIdx.x * blockDim.x + threadIdx.x;
  if (idx >= (E + N) * HEADS) return;
  int e = idx >> 2, hh = idx & 3;
  int s, d;
  if (e < E) {
    if (t >= 0 && etyp[e] != t) return;
    s = src[e]; d = dst[e];
  } else { s = d = e - E; }
  float val = es[s * HEADS + hh] + ed[d * HEADS + hh];
  val = val > 0.f ? val : 0.2f * val;          // leaky_relu(0.2)
  atomicMax(&menc[d * HEADS + hh], encf(val));
}

// ---------- softmax numerator scatter: s[dst,h]+=p ; num[dst,h,:]+=p*h[src,h,:] ----------
__global__ void edge_accum_kernel(const int* __restrict__ src, const int* __restrict__ dst,
                                  const int* __restrict__ etyp, int t,
                                  const float* __restrict__ es, const float* __restrict__ ed,
                                  const unsigned* __restrict__ menc, const float* __restrict__ h,
                                  float* __restrict__ ssum, float* __restrict__ num,
                                  int E, int N) {
  long long idx = (long long)blockIdx.x * blockDim.x + threadIdx.x;
  long long total = (long long)(E + N) * (HEADS * 32);
  if (idx >= total) return;
  int c  = (int)(idx & 31);          // 4-float chunk within head dim
  int hh = (int)((idx >> 5) & 3);
  long long e = idx >> 7;
  int s, d;
  if (e < E) {
    if (t >= 0 && etyp[e] != t) return;
    s = src[e]; d = dst[e];
  } else { s = d = (int)(e - E); }
  float val = es[s * HEADS + hh] + ed[d * HEADS + hh];
  val = val > 0.f ? val : 0.2f * val;
  float m = decf(menc[d * HEADS + hh]);
  float p = __expf(val - m);
  if (c == 0) atomicAdd(&ssum[d * HEADS + hh], p);
  const float4 hv = *(const float4*)(h + (size_t)s * (HEADS * HID) + hh * HID + c * 4);
  float* np = num + (size_t)d * (HEADS * HID) + hh * HID + c * 4;
  atomicAdd(np + 0, p * hv.x);
  atomicAdd(np + 1, p * hv.y);
  atomicAdd(np + 2, p * hv.z);
  atomicAdd(np + 3, p * hv.w);
}

// ---------- head-mean / s + bias; either out += softmax(etw)[t]*v (f32) or out = relu(v) (bf16) ----------
__global__ void gat_finalize_kernel(const float* __restrict__ num, const float* __restrict__ ssum,
                                    const float* __restrict__ bias, const float* __restrict__ etw,
                                    int t, float* __restrict__ outF, __bf16* __restrict__ outB,
                                    int N) {
  int idx = blockIdx.x * blockDim.x + threadIdx.x;  // n*HID + d
  if (idx >= N * HID) return;
  int n = idx >> 7, d = idx & (HID - 1);
  float acc = 0.f;
#pragma unroll
  for (int hh = 0; hh < HEADS; ++hh)
    acc += num[(size_t)n * (HEADS * HID) + hh * HID + d] / (ssum[n * HEADS + hh] + 1e-16f);
  acc = acc * (1.f / HEADS) + bias[d];
  if (etw) {
    float e0 = etw[0], e1 = etw[1], e2 = etw[2], e3 = etw[3];
    float mx = fmaxf(fmaxf(e0, e1), fmaxf(e2, e3));
    float x0 = __expf(e0 - mx), x1 = __expf(e1 - mx), x2 = __expf(e2 - mx), x3 = __expf(e3 - mx);
    float w  = (t == 0 ? x0 : t == 1 ? x1 : t == 2 ? x2 : x3) / (x0 + x1 + x2 + x3);
    outF[idx] += w * acc;
  } else {
    outB[idx] = f2bf(fmaxf(acc, 0.f));
  }
}

// ---------- per-node 4-head cross attention (wave32 per node) ----------
__global__ void cross_attn_kernel(const float* __restrict__ q, const float* __restrict__ k,
                                  const float* __restrict__ v, float* __restrict__ sem, int N) {
  int wpb  = blockDim.x >> 5;
  int node = blockIdx.x * wpb + (threadIdx.x >> 5);
  if (node >= N) return;
  int lane = threadIdx.x & 31;
  size_t base = (size_t)node * HID;
  float qv[HEADS], kv[HEADS], vv[HEADS];
#pragma unroll
  for (int h = 0; h < HEADS; ++h) {
    qv[h] = q[base + h * HD + lane];
    kv[h] = k[base + h * HD + lane];
    vv[h] = v[base + h * HD + lane];
  }
  float sc[HEADS][HEADS];
#pragma unroll
  for (int h = 0; h < HEADS; ++h)
#pragma unroll
    for (int g = 0; g < HEADS; ++g) {
      float p = qv[h] * kv[g];
#pragma unroll
      for (int off = 16; off > 0; off >>= 1) p += __shfl_xor(p, off, 32);
      sc[h][g] = p * 0.1767766952966369f;   // 1/sqrt(32)
    }
#pragma unroll
  for (int h = 0; h < HEADS; ++h) {
    float mx = fmaxf(fmaxf(sc[h][0], sc[h][1]), fmaxf(sc[h][2], sc[h][3]));
    float ex[HEADS], ss = 0.f, o = 0.f;
#pragma unroll
    for (int g = 0; g < HEADS; ++g) { ex[g] = __expf(sc[h][g] - mx); ss += ex[g]; }
#pragma unroll
    for (int g = 0; g < HEADS; ++g) o += ex[g] * vv[g];
    sem[base + h * HD + lane] = o / ss;
  }
}

// ---------- build bf16 concat [xs || bf16(sem)] ----------
__global__ void concat_kernel(const __bf16* __restrict__ xsb, const float* __restrict__ sem,
                              __bf16* __restrict__ cat, int N) {
  int idx = blockIdx.x * blockDim.x + threadIdx.x;
  if (idx >= N * HID) return;
  int n = idx >> 7, d = idx & (HID - 1);
  cat[(size_t)n * (2 * HID) + d]       = xsb[idx];
  cat[(size_t)n * (2 * HID) + HID + d] = f2bf(sem[idx]);
}

// ---------- LayerNorm + relu (wave32 per node) ----------
__global__ void ln_relu_kernel(const float* __restrict__ z, const float* __restrict__ g,
                               const float* __restrict__ b, float* __restrict__ out, int N) {
  int wpb  = blockDim.x >> 5;
  int node = blockIdx.x * wpb + (threadIdx.x >> 5);
  if (node >= N) return;
  int lane = threadIdx.x & 31;
  size_t base = (size_t)node * HID;
  float vals[4], s = 0.f, s2 = 0.f;
#pragma unroll
  for (int j = 0; j < 4; ++j) { float t = z[base + j * 32 + lane]; vals[j] = t; s += t; s2 += t * t; }
#pragma unroll
  for (int off = 16; off > 0; off >>= 1) { s += __shfl_xor(s, off, 32); s2 += __shfl_xor(s2, off, 32); }
  float mean = s * (1.f / HID);
  float var  = s2 * (1.f / HID) - mean * mean;
  float inv  = rsqrtf(var + 1e-5f);
#pragma unroll
  for (int j = 0; j < 4; ++j) {
    int d = j * 32 + lane;
    float t = (vals[j] - mean) * inv * g[d] + b[d];
    out[base + d] = fmaxf(t, 0.f);
  }
}

// ---------- host orchestration ----------
extern "C" void kernel_launch(void* const* d_in, const int* in_sizes, int n_in,
                              void* d_out, int out_size, void* d_ws, size_t ws_size,
                              hipStream_t stream) {
  const float* x    = (const float*)d_in[0];
  const float* cb   = (const float*)d_in[1];
  const int*   eidx = (const int*)d_in[2];
  const int*   eatt = (const int*)d_in[3];
  const float* Wt   = (const float*)d_in[4];
  const float* astp = (const float*)d_in[5];
  const float* adtp = (const float*)d_in[6];
  const float* btp  = (const float*)d_in[7];
  const float* etw  = (const float*)d_in[8];
  const float* Wg   = (const float*)d_in[9];
  const float* asg  = (const float*)d_in[10];
  const float* adg  = (const float*)d_in[11];
  const float* bg   = (const float*)d_in[12];
  const float* Wq   = (const float*)d_in[13];
  const float* bq   = (const float*)d_in[14];
  const float* Wk   = (const float*)d_in[15];
  const float* bk   = (const float*)d_in[16];
  const float* Wv   = (const float*)d_in[17];
  const float* bv   = (const float*)d_in[18];
  const float* Wf   = (const float*)d_in[19];
  const float* bfv  = (const float*)d_in[20];
  const float* lng  = (const float*)d_in[21];
  const float* lnb  = (const float*)d_in[22];

  const int N   = in_sizes[0] / HID;     // 50000
  const int E   = in_sizes[3];           // 320000
  const int CBd = in_sizes[1] / N;       // 768
  const int NT  = in_sizes[8];           // 4
  const int* src = eidx;
  const int* dst = eidx + E;

  // ---- workspace layout: f32 region then bf16 region ----
  float*    bufH   = (float*)d_ws;                          // [N, 512]
  float*    bufNum = bufH   + (size_t)N * 512;              // [N, 512]
  float*    bufXW  = bufNum + (size_t)N * 512;              // [N, 128] f32 accumulator
  float*    bufES  = bufXW  + (size_t)N * HID;              // [N, 4]
  float*    bufED  = bufES  + (size_t)N * HEADS;            // [N, 4]
  float*    bufS   = bufED  + (size_t)N * HEADS;            // [N, 4]
  unsigned* bufM   = (unsigned*)(bufS + (size_t)N * HEADS); // [N, 4]
  __bf16*   bfBase = (__bf16*)(bufM + (size_t)N * HEADS);
  __bf16* xbf   = bfBase;                                   // [N, 128]
  __bf16* cbbf  = xbf   + (size_t)N * HID;                  // [N, 768]
  __bf16* xwbf  = cbbf  + (size_t)N * CBd;                  // [N, 128]
  __bf16* xsbf  = xwbf  + (size_t)N * HID;                  // [N, 128]
  __bf16* catbf = xsbf  + (size_t)N * HID;                  // [N, 256]
  __bf16* WtT   = catbf + (size_t)N * (2 * HID);            // [NT, 512, 128]
  __bf16* WgT   = WtT + (size_t)NT * (HEADS * HID) * HID;   // [512, 128]
  __bf16* WqT   = WgT + (size_t)(HEADS * HID) * HID;        // [128, 128]
  __bf16* WkT   = WqT + (size_t)HID * HID;                  // [128, 768]
  __bf16* WvT   = WkT + (size_t)HID * CBd;                  // [128, 768]
  __bf16* WfT   = WvT + (size_t)HID * CBd;                  // [128, 256]
  // phase-3 f32 aliases (bufH/bufNum free after GAT stages)
  float* bufQ   = bufH;
  float* bufK   = bufH + (size_t)N * HID;
  float* bufV   = bufH + 2 * (size_t)N * HID;
  float* bufSem = bufH + 3 * (size_t)N * HID;
  float* bufZ   = bufNum;                                   // [N, 128]

  auto gemm = [&](const __bf16* A, const __bf16* BT, const float* bias, float* C,
                  int M, int Nc, int K) {
    int waves  = (M >> 4) * (Nc >> 6);
    int blocks = (waves + 7) / 8;
    wmma_gemm_bf16<<<dim3(blocks), dim3(256), 0, stream>>>(A, BT, bias, C, M, Nc, K);
  };
  auto cvt = [&](const float* s, __bf16* d, long long n) {
    int blocks = (int)((n / 4 + 255) / 256);
    cvt_bf16_kernel<<<dim3(blocks), dim3(256), 0, stream>>>(s, d, n);
  };
  auto cvtT = [&](const float* W, __bf16* WTp, int K, int Nc) {
    int blocks = (K * Nc + 255) / 256;
    cvt_tr_kernel<<<dim3(blocks), dim3(256), 0, stream>>>(W, WTp, K, Nc);
  };

  const int blkME   = ((E + N) * HEADS + 255) / 256;
  const long long totAC = (long long)(E + N) * (HEADS * 32);
  const int blkAC   = (int)((totAC + 255) / 256);
  const int blkND   = (N * HID + 255) / 256;
  const int blkDots = (N * HEADS + 255) / 256;
  const int blkWave = (N + 7) / 8;

  // ---- one-time conversions ----
  cvt(x,  xbf,  (long long)N * HID);
  cvt(cb, cbbf, (long long)N * CBd);
  for (int t = 0; t < NT; ++t)
    cvtT(Wt + (size_t)t * HID * (HEADS * HID), WtT + (size_t)t * (HEADS * HID) * HID,
         HID, HEADS * HID);
  cvtT(Wg, WgT, HID, HEADS * HID);
  cvtT(Wq, WqT, HID, HID);
  cvtT(Wk, WkT, CBd, HID);
  cvtT(Wv, WvT, CBd, HID);
  cvtT(Wf, WfT, 2 * HID, HID);

  hipMemsetAsync(bufXW, 0, (size_t)N * HID * sizeof(float), stream);

  // ---- per-edge-type GATConvs, routing-weighted accumulation into xw ----
  for (int t = 0; t < NT; ++t) {
    gemm(xbf, WtT + (size_t)t * (HEADS * HID) * HID, nullptr, bufH, N, HEADS * HID, HID);
    node_dots<<<blkDots, 256, 0, stream>>>(bufH, astp + (size_t)t * HEADS * HID,
                                           adtp + (size_t)t * HEADS * HID, bufES, bufED, N);
    hipMemsetAsync(bufM, 0, (size_t)N * HEADS * sizeof(unsigned), stream);
    hipMemsetAsync(bufS, 0, (size_t)N * HEADS * sizeof(float), stream);
    hipMemsetAsync(bufNum, 0, (size_t)N * 512 * sizeof(float), stream);
    edge_max_kernel<<<blkME, 256, 0, stream>>>(src, dst, eatt, t, bufES, bufED, bufM, E, N);
    edge_accum_kernel<<<blkAC, 256, 0, stream>>>(src, dst, eatt, t, bufES, bufED, bufM,
                                                 bufH, bufS, bufNum, E, N);
    gat_finalize_kernel<<<blkND, 256, 0, stream>>>(bufNum, bufS, btp + (size_t)t * HID,
                                                   etw, t, bufXW, nullptr, N);
  }
  cvt(bufXW, xwbf, (long long)N * HID);

  // ---- fusion GATConv (all edges) + relu -> xs (bf16) ----
  gemm(xwbf, WgT, nullptr, bufH, N, HEADS * HID, HID);
  node_dots<<<blkDots, 256, 0, stream>>>(bufH, asg, adg, bufES, bufED, N);
  hipMemsetAsync(bufM, 0, (size_t)N * HEADS * sizeof(unsigned), stream);
  hipMemsetAsync(bufS, 0, (size_t)N * HEADS * sizeof(float), stream);
  hipMemsetAsync(bufNum, 0, (size_t)N * 512 * sizeof(float), stream);
  edge_max_kernel<<<blkME, 256, 0, stream>>>(src, dst, eatt, -1, bufES, bufED, bufM, E, N);
  edge_accum_kernel<<<blkAC, 256, 0, stream>>>(src, dst, eatt, -1, bufES, bufED, bufM,
                                               bufH, bufS, bufNum, E, N);
  gat_finalize_kernel<<<blkND, 256, 0, stream>>>(bufNum, bufS, bg, nullptr, -1,
                                                 nullptr, xsbf, N);

  // ---- cross attention with CodeBERT ----
  gemm(xsbf, WqT, bq, bufQ, N, HID, HID);
  gemm(cbbf, WkT, bk, bufK, N, HID, CBd);
  gemm(cbbf, WvT, bv, bufV, N, HID, CBd);
  cross_attn_kernel<<<blkWave, 256, 0, stream>>>(bufQ, bufK, bufV, bufSem, N);

  // ---- fusion MLP + LayerNorm + relu ----
  concat_kernel<<<blkND, 256, 0, stream>>>(xsbf, bufSem, catbf, N);
  gemm(catbf, WfT, bfv, bufZ, N, HID, 2 * HID);
  ln_relu_kernel<<<blkWave, 256, 0, stream>>>(bufZ, lng, lnb, (float*)d_out, N);
}
